// DecoderBlock_30803505447075
// MI455X (gfx1250) — matrix operand order repaired and checked by reading, not compiled, not run on hardware
//
#include <hip/hip_runtime.h>

// ---------------------------------------------------------------------------
// Types for CDNA5 WMMA (wave32)
// ---------------------------------------------------------------------------
typedef __attribute__((ext_vector_type(16))) __bf16       v16bf;
typedef __attribute__((ext_vector_type(8)))  float        v8f;
typedef __attribute__((ext_vector_type(4)))  unsigned int uint4v;

static __device__ inline v8f vzero8() {
  v8f z = {0.f, 0.f, 0.f, 0.f, 0.f, 0.f, 0.f, 0.f};
  return z;
}

static __device__ inline v8f wmma_bf16(v16bf a, v16bf b, v8f c) {
  // D(f32 16x16) = A(bf16 16x32) * B(bf16 32x16) + C
  return __builtin_amdgcn_wmma_f32_16x16x32_bf16(
      /*neg_a=*/false, a, /*neg_b=*/false, b,
      /*c_mod=*/(short)0, c, /*reuse_a=*/false, /*reuse_b=*/false);
}

// A fragment (16x32 bf16). LDS tile is row-major [M][K], `stride` elements/row.
static __device__ inline v16bf load_a_frag(const __bf16* base, int stride,
                                           int row0, int laneM, int hi) {
  union { unsigned int u[8]; v16bf v; } f;
  const __bf16* rp = base + (row0 + laneM) * stride;
#pragma unroll
  for (int j = 0; j < 8; ++j) {
    int kb = (j < 4 ? 2 * j : 2 * j + 8) + hi * 8;
    f.u[j] = *(const unsigned int*)(rp + kb);
  }
  return f.v;
}

// B fragment (K=32 x N=16). LDS tile is row-major [N][K], `stride` elements/row.
static __device__ inline v16bf load_b_frag(const __bf16* base, int stride,
                                           int n0, int laneM, int hi, int koff) {
  union { unsigned int u[8]; v16bf v; } f;
  const __bf16* rp = base + (n0 + laneM) * stride + koff + hi * 8;
#pragma unroll
  for (int j = 0; j < 8; ++j) f.u[j] = *(const unsigned int*)(rp + (j < 4 ? 2 * j : 2 * j + 8));
  return f.v;
}

// ---------------------------------------------------------------------------
// Elementwise converts
// ---------------------------------------------------------------------------
__global__ void conv_bf16_kernel(const float* __restrict__ in,
                                 __bf16* __restrict__ out) {
  int i = blockIdx.x * 256 + threadIdx.x;
  out[i] = (__bf16)in[i];
}

// f32 [K,N] -> bf16 [N,K]  (transpose so GEMM B tiles are contiguous along K)
__global__ void conv_t_kernel(const float* __restrict__ in,
                              __bf16* __restrict__ out, int K, int N) {
  int idx = blockIdx.x * 256 + threadIdx.x;   // < K*N <= 4M
  int k = idx / N;
  int n = idx - k * N;
  out[(size_t)n * K + k] = (__bf16)in[idx];
}

// ---------------------------------------------------------------------------
// LayerNorm: f32 [rows, D=1024] -> bf16, one block (256 thr) per row
// ---------------------------------------------------------------------------
__global__ __launch_bounds__(256) void ln_bf16_kernel(
    const float* __restrict__ x, const float* __restrict__ w,
    const float* __restrict__ b, __bf16* __restrict__ out, int D) {
  __shared__ float red[256];
  const int tid = threadIdx.x;
  const float* xr = x + (size_t)blockIdx.x * D;
  float4 v = *(const float4*)(xr + tid * 4);

  float s = v.x + v.y + v.z + v.w;
  red[tid] = s;
  __syncthreads();
#pragma unroll
  for (int o = 128; o > 0; o >>= 1) {
    if (tid < o) red[tid] += red[tid + o];
    __syncthreads();
  }
  float mean = red[0] / (float)D;
  __syncthreads();

  float d0 = v.x - mean, d1 = v.y - mean, d2 = v.z - mean, d3 = v.w - mean;
  red[tid] = d0 * d0 + d1 * d1 + d2 * d2 + d3 * d3;
  __syncthreads();
#pragma unroll
  for (int o = 128; o > 0; o >>= 1) {
    if (tid < o) red[tid] += red[tid + o];
    __syncthreads();
  }
  float inv = rsqrtf(red[0] / (float)D + 1e-5f);

  const float4 wv = *(const float4*)(w + tid * 4);
  const float4 bv = *(const float4*)(b + tid * 4);
  __bf16* op = out + (size_t)blockIdx.x * D + tid * 4;
  op[0] = (__bf16)(d0 * inv * wv.x + bv.x);
  op[1] = (__bf16)(d1 * inv * wv.y + bv.y);
  op[2] = (__bf16)(d2 * inv * wv.z + bv.z);
  op[3] = (__bf16)(d3 * inv * wv.w + bv.w);
}

// ---------------------------------------------------------------------------
// WMMA GEMM: C[M,N] = A[M,K](bf16) @ Bt[N,K](bf16)^T + bias (+residual)(+relu)
// Block 256 thr = 8 waves, tile 64(M) x 128(N), K chunks of 32.
// Waves in 2x4 grid; each wave computes a 32x32 patch = 2x2 WMMA tiles
// (4 WMMAs per chunk from 2 A-frags x 2 B-frags).
// LDS ping-pong: one barrier per K-chunk; unconditional prefetch in the
// steady-state loop (last chunk peeled) so no zero-fills/branches in the loop.
// ---------------------------------------------------------------------------
#define GEMM_KC   32
#define GEMM_STR  40   // padded LDS row stride (bf16 elems); 80B keeps 16B align

__global__ __launch_bounds__(256) void gemm_bf16_kernel(
    const __bf16* __restrict__ A, const __bf16* __restrict__ Bt,
    const float* __restrict__ bias, const float* __restrict__ residual,
    float* __restrict__ outF, __bf16* __restrict__ outB,
    int M, int N, int K, int relu) {
  __shared__ __bf16 As[2][64 * GEMM_STR];
  __shared__ __bf16 Bs[2][128 * GEMM_STR];

  const int tid   = threadIdx.x;
  const int lane  = tid & 31;
  const int laneM = lane & 15;
  const int hi    = lane >> 4;
  const int wave  = tid >> 5;
  const int wm    = wave >> 2;   // 0..1 -> rows wm*32..+31
  const int wn    = wave & 3;    // 0..3 -> cols wn*32..+31
  const int m0    = blockIdx.y * 64;
  const int n0    = blockIdx.x * 128;

  // cooperative staging indices
  const int lrA = tid >> 2;        // 0..63
  const int lcA = (tid & 3) * 8;   // 0,8,16,24
  const int lrB = tid >> 1;        // 0..127
  const int lcB = (tid & 1) * 16;  // 0,16

  const __bf16* Aptr = A  + (size_t)(m0 + lrA) * K + lcA;
  const __bf16* Bptr = Bt + (size_t)(n0 + lrB) * K + lcB;

  v8f acc[2][2] = {{vzero8(), vzero8()}, {vzero8(), vzero8()}};

  // prologue: stage chunk 0 into buffer 0
  {
    uint4v av  = *(const uint4v*)(Aptr);
    uint4v bv0 = *(const uint4v*)(Bptr);
    uint4v bv1 = *(const uint4v*)(Bptr + 8);
    *(uint4v*)(&As[0][lrA * GEMM_STR + lcA])     = av;
    *(uint4v*)(&Bs[0][lrB * GEMM_STR + lcB])     = bv0;
    *(uint4v*)(&Bs[0][lrB * GEMM_STR + lcB + 8]) = bv1;
  }
  __syncthreads();

  // steady state: unconditional prefetch of k0+KC, compute k0, one barrier
  int buf = 0;
  for (int k0 = 0; k0 + GEMM_KC < K; k0 += GEMM_KC) {
    uint4v av  = *(const uint4v*)(Aptr + k0 + GEMM_KC);
    uint4v bv0 = *(const uint4v*)(Bptr + k0 + GEMM_KC);
    uint4v bv1 = *(const uint4v*)(Bptr + k0 + GEMM_KC + 8);

    v16bf a0 = load_a_frag(&As[buf][0], GEMM_STR, wm * 32,      laneM, hi);
    v16bf a1 = load_a_frag(&As[buf][0], GEMM_STR, wm * 32 + 16, laneM, hi);
    v16bf b0 = load_b_frag(&Bs[buf][0], GEMM_STR, wn * 32,      laneM, hi, 0);
    v16bf b1 = load_b_frag(&Bs[buf][0], GEMM_STR, wn * 32 + 16, laneM, hi, 0);
    acc[0][0] = wmma_bf16(a0, b0, acc[0][0]);
    acc[0][1] = wmma_bf16(a0, b1, acc[0][1]);
    acc[1][0] = wmma_bf16(a1, b0, acc[1][0]);
    acc[1][1] = wmma_bf16(a1, b1, acc[1][1]);

    // store prefetch into the other buffer (no conflict with current readers)
    *(uint4v*)(&As[buf ^ 1][lrA * GEMM_STR + lcA])     = av;
    *(uint4v*)(&Bs[buf ^ 1][lrB * GEMM_STR + lcB])     = bv0;
    *(uint4v*)(&Bs[buf ^ 1][lrB * GEMM_STR + lcB + 8]) = bv1;
    __syncthreads();
    buf ^= 1;
  }

  // final chunk: compute only (no prefetch, no barrier needed afterwards)
  {
    v16bf a0 = load_a_frag(&As[buf][0], GEMM_STR, wm * 32,      laneM, hi);
    v16bf a1 = load_a_frag(&As[buf][0], GEMM_STR, wm * 32 + 16, laneM, hi);
    v16bf b0 = load_b_frag(&Bs[buf][0], GEMM_STR, wn * 32,      laneM, hi, 0);
    v16bf b1 = load_b_frag(&Bs[buf][0], GEMM_STR, wn * 32 + 16, laneM, hi, 0);
    acc[0][0] = wmma_bf16(a0, b0, acc[0][0]);
    acc[0][1] = wmma_bf16(a0, b1, acc[0][1]);
    acc[1][0] = wmma_bf16(a1, b0, acc[1][0]);
    acc[1][1] = wmma_bf16(a1, b1, acc[1][1]);
  }

#pragma unroll
  for (int mt = 0; mt < 2; ++mt) {
#pragma unroll
    for (int nt = 0; nt < 2; ++nt) {
      const int col  = n0 + wn * 32 + nt * 16 + laneM;
      const float bc = bias ? bias[col] : 0.f;
#pragma unroll
      for (int r = 0; r < 8; ++r) {
        int gm = m0 + wm * 32 + mt * 16 + r + hi * 8;
        float v = acc[mt][nt][r] + bc;
        if (residual) v += residual[(size_t)gm * N + col];
        if (relu) v = v > 0.f ? v : 0.f;
        if (outF) outF[(size_t)gm * N + col] = v;
        if (outB) outB[(size_t)gm * N + col] = (__bf16)v;
      }
    }
  }
}

// ---------------------------------------------------------------------------
// Flash attention (bf16 in/out, f32 accum). HD=64, heads on grid.y.
// Block 128 thr = 4 waves; q tile 64 rows (16 per wave); kv tiles of 64.
// ---------------------------------------------------------------------------
#define AHD   64
#define ASTRD 72   // padded LDS stride (144B, 16B-aligned)

__global__ __launch_bounds__(128) void attn_bf16_kernel(
    const __bf16* __restrict__ Q, const __bf16* __restrict__ Km,
    const __bf16* __restrict__ Vm, __bf16* __restrict__ O,
    int DQtot, int Lk, int causal) {
  __shared__ __bf16 Ks[64 * ASTRD];       // [kpos][dim]  (B layout for Q·K^T)
  __shared__ __bf16 Vs[64 * ASTRD];       // [dim][kpos]  (B layout for P·V)
  __shared__ __bf16 Ps[4 * 16 * ASTRD];   // per-wave P tile, [row][kpos]

  const int tid   = threadIdx.x;
  const int lane  = tid & 31;
  const int laneM = lane & 15;
  const int hi    = lane >> 4;
  const int wave  = tid >> 5;            // 0..3
  const int qb    = blockIdx.x;          // q tile of 64
  const int hd0   = blockIdx.y * AHD;
  const int qr0   = qb * 64 + wave * 16; // this wave's first q row

  // Q fragments (A layout), loaded once from global
  v16bf qf[2];
  {
    const __bf16* qp = Q + (size_t)(qr0 + laneM) * DQtot + hd0;
#pragma unroll
    for (int c = 0; c < 2; ++c) {
      union { unsigned int u[8]; v16bf v; } f;
#pragma unroll
      for (int j = 0; j < 8; ++j) {
        int kb = (j < 4 ? 2 * j : 2 * j + 8) + hi * 8 + 32 * c;
        f.u[j] = *(const unsigned int*)(qp + kb);
      }
      qf[c] = f.v;
    }
  }

  float mrow[8], lrow[8];
#pragma unroll
  for (int r = 0; r < 8; ++r) { mrow[r] = -1e30f; lrow[r] = 0.f; }
  v8f acc[4] = {vzero8(), vzero8(), vzero8(), vzero8()};

  const int jmax = causal ? (qb + 1) : (Lk / 64);
  for (int j = 0; j < jmax; ++j) {
    __syncthreads();
    // stage K [kpos][dim] and V^T [dim][kpos]
#pragma unroll
    for (int i = 0; i < 4; ++i) {
      int cc = tid + 128 * i;          // 0..511 chunks of 8 bf16
      int r  = cc >> 3;
      int c  = (cc & 7) * 8;
      uint4v kv = *(const uint4v*)(Km + (size_t)(j * 64 + r) * DQtot + hd0 + c);
      *(uint4v*)(Ks + r * ASTRD + c) = kv;
      const __bf16* vp = Vm + (size_t)(j * 64 + r) * DQtot + hd0 + c;
#pragma unroll
      for (int u = 0; u < 8; ++u) Vs[(c + u) * ASTRD + r] = vp[u];
    }
    __syncthreads();

    // scores: 4 tiles of 16 kpos each, K-dim = HD via 2 WMMAs
    float p[4][8];
#pragma unroll
    for (int t = 0; t < 4; ++t) {
      v8f sacc = vzero8();
      v16bf b0 = load_b_frag(Ks, ASTRD, t * 16, laneM, hi, 0);
      v16bf b1 = load_b_frag(Ks, ASTRD, t * 16, laneM, hi, 32);
      sacc = wmma_bf16(qf[0], b0, sacc);
      sacc = wmma_bf16(qf[1], b1, sacc);
#pragma unroll
      for (int r = 0; r < 8; ++r) {
        float s = sacc[r] * 0.125f;   // 1/sqrt(64)
        if (causal && j == qb) {
          int qpos = qr0 + r + 8 * hi;
          int kpos = j * 64 + t * 16 + laneM;
          if (kpos > qpos) s = -1e30f;
        }
        p[t][r] = s;
      }
    }

    // online softmax (rows live across 16-lane halves in C layout)
    float mnew[8];
#pragma unroll
    for (int r = 0; r < 8; ++r) {
      float mx = p[0][r];
#pragma unroll
      for (int t = 1; t < 4; ++t) mx = fmaxf(mx, p[t][r]);
#pragma unroll
      for (int off = 8; off >= 1; off >>= 1)
        mx = fmaxf(mx, __shfl_xor(mx, off, 32));
      mnew[r] = fmaxf(mrow[r], mx);
    }
#pragma unroll
    for (int r = 0; r < 8; ++r) {
      float alpha = __expf(mrow[r] - mnew[r]);
      float rs = 0.f;
#pragma unroll
      for (int t = 0; t < 4; ++t) {
        float e = __expf(p[t][r] - mnew[r]);
        p[t][r] = e;
        rs += e;
      }
#pragma unroll
      for (int off = 8; off >= 1; off >>= 1) rs += __shfl_xor(rs, off, 32);
      lrow[r] = lrow[r] * alpha + rs;
      mrow[r] = mnew[r];
#pragma unroll
      for (int t = 0; t < 4; ++t) acc[t][r] *= alpha;
    }

    // C-layout f32 -> A-layout bf16 via LDS
    {
      __bf16* pw = Ps + wave * 16 * ASTRD;
#pragma unroll
      for (int t = 0; t < 4; ++t)
#pragma unroll
        for (int r = 0; r < 8; ++r)
          pw[(r + 8 * hi) * ASTRD + t * 16 + laneM] = (__bf16)p[t][r];
    }
    __syncthreads();

    v16bf pf[2];
    {
      const __bf16* pp = Ps + (wave * 16 + laneM) * ASTRD;
#pragma unroll
      for (int c = 0; c < 2; ++c) {
        union { unsigned int u[8]; v16bf v; } f;
#pragma unroll
        for (int jj = 0; jj < 8; ++jj) {
          int kb = (jj < 4 ? 2 * jj : 2 * jj + 8) + hi * 8 + 32 * c;
          f.u[jj] = *(const unsigned int*)(pp + kb);
        }
        pf[c] = f.v;
      }
    }
#pragma unroll
    for (int t = 0; t < 4; ++t) {
      v16bf v0 = load_b_frag(Vs, ASTRD, t * 16, laneM, hi, 0);
      v16bf v1 = load_b_frag(Vs, ASTRD, t * 16, laneM, hi, 32);
      acc[t] = wmma_bf16(pf[0], v0, acc[t]);
      acc[t] = wmma_bf16(pf[1], v1, acc[t]);
    }
  }

  // epilogue: divide by row sum, write bf16
#pragma unroll
  for (int t = 0; t < 4; ++t)
#pragma unroll
    for (int r = 0; r < 8; ++r) {
      int q = qr0 + r + 8 * hi;
      float v = acc[t][r] / lrow[r];
      O[(size_t)q * DQtot + hd0 + t * 16 + laneM] = (__bf16)v;
    }
}

// ---------------------------------------------------------------------------
// Host orchestration
// ---------------------------------------------------------------------------
extern "C" void kernel_launch(void* const* d_in, const int* in_sizes, int n_in,
                              void* d_out, int out_size, void* d_ws, size_t ws_size,
                              hipStream_t stream) {
  (void)in_sizes; (void)n_in; (void)out_size; (void)ws_size;
  const int S = 2048, T = 2048, D = 1024, H = 16, DQ = 1024, F = 4096;

  const float* inputs = (const float*)d_in[0];
  const float* enc    = (const float*)d_in[1];
  const float* ln1w = (const float*)d_in[4],  *ln1b = (const float*)d_in[5];
  const float* ln2w = (const float*)d_in[6],  *ln2b = (const float*)d_in[7];
  const float* ln3w = (const float*)d_in[8],  *ln3b = (const float*)d_in[9];

  char* ws = (char*)d_ws;
  size_t off = 0;
  auto alloc = [&](size_t bytes) {
    size_t o = off;
    off += (bytes + 255) & ~(size_t)255;
    return (void*)(ws + o);
  };

  __bf16* nbuf = (__bf16*)alloc((size_t)S * D * 2);
  __bf16* encb = (__bf16*)alloc((size_t)T * D * 2);
  __bf16* qbuf = (__bf16*)alloc((size_t)S * DQ * 2);
  __bf16* kbuf = (__bf16*)alloc((size_t)T * DQ * 2);
  __bf16* vbuf = (__bf16*)alloc((size_t)T * DQ * 2);
  __bf16* ctxb = (__bf16*)alloc((size_t)S * DQ * 2);
  __bf16* hbuf = (__bf16*)alloc((size_t)S * F * 2);
  float*  x1   = (float*)alloc((size_t)S * D * 4);
  float*  x2   = (float*)alloc((size_t)S * D * 4);
  __bf16* saWqT = (__bf16*)alloc((size_t)DQ * D * 2);
  __bf16* saWkT = (__bf16*)alloc((size_t)DQ * D * 2);
  __bf16* saWvT = (__bf16*)alloc((size_t)DQ * D * 2);
  __bf16* saWoT = (__bf16*)alloc((size_t)D * DQ * 2);
  __bf16* caWqT = (__bf16*)alloc((size_t)DQ * D * 2);
  __bf16* caWkT = (__bf16*)alloc((size_t)DQ * D * 2);
  __bf16* caWvT = (__bf16*)alloc((size_t)DQ * D * 2);
  __bf16* caWoT = (__bf16*)alloc((size_t)D * DQ * 2);
  __bf16* wiT   = (__bf16*)alloc((size_t)F * D * 2);
  __bf16* woT   = (__bf16*)alloc((size_t)D * F * 2);

  dim3 b256(256), b128(128);

  // weight convert+transpose (f32 [K,N] -> bf16 [N,K])
  conv_t_kernel<<<dim3((D * DQ) / 256), b256, 0, stream>>>((const float*)d_in[10], saWqT, D, DQ);
  conv_t_kernel<<<dim3((D * DQ) / 256), b256, 0, stream>>>((const float*)d_in[12], saWkT, D, DQ);
  conv_t_kernel<<<dim3((D * DQ) / 256), b256, 0, stream>>>((const float*)d_in[14], saWvT, D, DQ);
  conv_t_kernel<<<dim3((DQ * D) / 256), b256, 0, stream>>>((const float*)d_in[16], saWoT, DQ, D);
  conv_t_kernel<<<dim3((D * DQ) / 256), b256, 0, stream>>>((const float*)d_in[18], caWqT, D, DQ);
  conv_t_kernel<<<dim3((D * DQ) / 256), b256, 0, stream>>>((const float*)d_in[20], caWkT, D, DQ);
  conv_t_kernel<<<dim3((D * DQ) / 256), b256, 0, stream>>>((const float*)d_in[22], caWvT, D, DQ);
  conv_t_kernel<<<dim3((DQ * D) / 256), b256, 0, stream>>>((const float*)d_in[24], caWoT, DQ, D);
  conv_t_kernel<<<dim3((D * F) / 256),  b256, 0, stream>>>((const float*)d_in[26], wiT,  D, F);
  conv_t_kernel<<<dim3((F * D) / 256),  b256, 0, stream>>>((const float*)d_in[28], woT,  F, D);
  conv_bf16_kernel<<<dim3((T * D) / 256), b256, 0, stream>>>(enc, encb);

  // ---- self attention ----
  ln_bf16_kernel<<<dim3(S), b256, 0, stream>>>(inputs, ln1w, ln1b, nbuf, D);
  gemm_bf16_kernel<<<dim3(DQ / 128, S / 64), b256, 0, stream>>>(
      nbuf, saWqT, (const float*)d_in[11], nullptr, nullptr, qbuf, S, DQ, D, 0);
  gemm_bf16_kernel<<<dim3(DQ / 128, S / 64), b256, 0, stream>>>(
      nbuf, saWkT, (const float*)d_in[13], nullptr, nullptr, kbuf, S, DQ, D, 0);
  gemm_bf16_kernel<<<dim3(DQ / 128, S / 64), b256, 0, stream>>>(
      nbuf, saWvT, (const float*)d_in[15], nullptr, nullptr, vbuf, S, DQ, D, 0);
  attn_bf16_kernel<<<dim3(S / 64, H), b128, 0, stream>>>(qbuf, kbuf, vbuf, ctxb, DQ, S, 1);
  gemm_bf16_kernel<<<dim3(D / 128, S / 64), b256, 0, stream>>>(
      ctxb, saWoT, (const float*)d_in[17], inputs, x1, nullptr, S, D, DQ, 0);

  // ---- cross attention ----
  ln_bf16_kernel<<<dim3(S), b256, 0, stream>>>(x1, ln2w, ln2b, nbuf, D);
  gemm_bf16_kernel<<<dim3(DQ / 128, S / 64), b256, 0, stream>>>(
      nbuf, caWqT, (const float*)d_in[19], nullptr, nullptr, qbuf, S, DQ, D, 0);
  gemm_bf16_kernel<<<dim3(DQ / 128, T / 64), b256, 0, stream>>>(
      encb, caWkT, (const float*)d_in[21], nullptr, nullptr, kbuf, T, DQ, D, 0);
  gemm_bf16_kernel<<<dim3(DQ / 128, T / 64), b256, 0, stream>>>(
      encb, caWvT, (const float*)d_in[23], nullptr, nullptr, vbuf, T, DQ, D, 0);
  attn_bf16_kernel<<<dim3(S / 64, H), b128, 0, stream>>>(qbuf, kbuf, vbuf, ctxb, DQ, T, 0);
  gemm_bf16_kernel<<<dim3(D / 128, S / 64), b256, 0, stream>>>(
      ctxb, caWoT, (const float*)d_in[25], x1, x2, nullptr, S, D, DQ, 0);

  // ---- MLP ----
  ln_bf16_kernel<<<dim3(S), b256, 0, stream>>>(x2, ln3w, ln3b, nbuf, D);
  gemm_bf16_kernel<<<dim3(F / 128, S / 64), b256, 0, stream>>>(
      nbuf, wiT, (const float*)d_in[27], nullptr, nullptr, hbuf, S, F, D, 1);
  gemm_bf16_kernel<<<dim3(D / 128, S / 64), b256, 0, stream>>>(
      hbuf, woT, (const float*)d_in[29], x2, (float*)d_out, nullptr, S, D, F, 0);
}
